// DenseDilatedKnnGraph_47863115547127
// MI455X (gfx1250) — compile-verified, matrix operand order checked by reading, not run on hardware
//
#include <hip/hip_runtime.h>

typedef __attribute__((ext_vector_type(16))) __bf16 v16bf;
typedef __attribute__((ext_vector_type(8)))  float  v8f;

#define B_    4
#define CIN   512
#define COUT  256
#define NPTS  4096
#define KNN   9
#define BN_EPS 1e-5f
#define NEG_INF (-3.402823e38f)

static __device__ __forceinline__ unsigned short f2bf(float f) {
  unsigned u = __float_as_uint(f);
  unsigned r = u + 0x7FFFu + ((u >> 16) & 1u);   // round-to-nearest-even
  return (unsigned short)(r >> 16);
}

static __device__ __forceinline__ v8f vzero8() {
  v8f z;
  #pragma unroll
  for (int i = 0; i < 8; ++i) z[i] = 0.f;
  return z;
}

// ---------------------------------------------------------------------------
// Prep: W -> bf16, zero BN partial sums
// ---------------------------------------------------------------------------
__global__ void __launch_bounds__(256) k_prep_w(const float* __restrict__ w,
                                                unsigned short* __restrict__ wbf,
                                                float* sums, float* sumsq) {
  int i = blockIdx.x * 256 + threadIdx.x;
  if (i < COUT * CIN) wbf[i] = f2bf(w[i]);
  if (i < COUT) { sums[i] = 0.f; sumsq[i] = 0.f; }
}

// x (B,512,4096) f32 -> xbfT (B,4096,512) bf16 (K-contiguous for GEMM B-operand)
__global__ void __launch_bounds__(256) k_xT(const float* __restrict__ x,
                                            unsigned short* __restrict__ xbfT) {
  long i = (long)blockIdx.x * 256 + threadIdx.x;   // output linear index
  int c = (int)(i % CIN);
  long t = i / CIN;
  int n = (int)(t % NPTS);
  int b = (int)(t / NPTS);
  xbfT[i] = f2bf(x[((long)(b * CIN + c)) * NPTS + n]);
}

// ---------------------------------------------------------------------------
// WMMA bf16 GEMM mainloop: C(128x128) += A(128xK) * Bt(128xK)^T
// A rows = output rows, Bt rows = output cols; both K-contiguous.
// Per CDNA5 ISA: lane l element j of a 16x32 bf16 fragment holds
// K = j + 8*((j>>3) + (l>>4)), i.e. two ds_load_b128 per fragment.
// ---------------------------------------------------------------------------
__device__ __forceinline__ void gemm_mainloop(
    const unsigned short* __restrict__ A, const unsigned short* __restrict__ Bt,
    int lda, int ldb, int K, int row0, int col0,
    unsigned short (*At)[40], unsigned short (*Bs)[40], v8f acc[4][2]) {
  int tid  = threadIdx.x;
  int wave = tid >> 5, lane = tid & 31;
  int wr = wave >> 2, wc = wave & 3;       // 2 x 4 wave grid -> 64x32 per wave
  int l16 = lane & 15, lhi = lane >> 4;

  for (int k0 = 0; k0 < K; k0 += 32) {
    #pragma unroll
    for (int i = 0; i < 2; ++i) {          // 512 x 16B chunks, 256 threads
      int cid = tid + i * 256;
      int r = cid >> 2, kc = (cid & 3) << 3;
      *(uint4*)&At[r][kc] = *(const uint4*)&A[(long)(row0 + r) * lda + k0 + kc];
      *(uint4*)&Bs[r][kc] = *(const uint4*)&Bt[(long)(col0 + r) * ldb + k0 + kc];
    }
    __syncthreads();

    union Frag { uint4 u[2]; v16bf v; };
    Frag bf2[2];
    #pragma unroll
    for (int fj = 0; fj < 2; ++fj) {
      int br = wc * 32 + fj * 16 + l16;
      bf2[fj].u[0] = *(const uint4*)&Bs[br][lhi * 8];
      bf2[fj].u[1] = *(const uint4*)&Bs[br][lhi * 8 + 16];
    }
    #pragma unroll
    for (int fi = 0; fi < 4; ++fi) {
      Frag af;
      int ar = wr * 64 + fi * 16 + l16;
      af.u[0] = *(const uint4*)&At[ar][lhi * 8];
      af.u[1] = *(const uint4*)&At[ar][lhi * 8 + 16];
      #pragma unroll
      for (int fj = 0; fj < 2; ++fj) {
        acc[fi][fj] = __builtin_amdgcn_wmma_f32_16x16x32_bf16(
            false, af.v, false, bf2[fj].v, (short)0, acc[fi][fj], false, false);
      }
    }
    __syncthreads();
  }
}

// ---------------------------------------------------------------------------
// Conv GEMM: Y(b, 256, 4096) = Wbf(256x512) @ X(512x4096) + bias, plus BN stats
// ---------------------------------------------------------------------------
__global__ void __launch_bounds__(256) k_conv(
    const unsigned short* __restrict__ wbf, const unsigned short* __restrict__ xbfT,
    const float* __restrict__ conv_b, float* __restrict__ y,
    float* sums, float* sumsq) {
  __shared__ unsigned short At[128][40];
  __shared__ unsigned short Bs[128][40];
  __shared__ float s_sum[128], s_ssq[128];

  int b = blockIdx.z;
  int row0 = blockIdx.y * 128;   // channel tile
  int col0 = blockIdx.x * 128;   // point tile
  v8f acc[4][2];
  #pragma unroll
  for (int i = 0; i < 4; ++i)
    #pragma unroll
    for (int j = 0; j < 2; ++j) acc[i][j] = vzero8();

  gemm_mainloop(wbf, xbfT + (long)b * NPTS * CIN, CIN, CIN, CIN, row0, col0, At, Bs, acc);

  if (threadIdx.x < 128) { s_sum[threadIdx.x] = 0.f; s_ssq[threadIdx.x] = 0.f; }
  __syncthreads();

  int wave = threadIdx.x >> 5, lane = threadIdx.x & 31;
  int wr = wave >> 2, wc = wave & 3, l16 = lane & 15, lhi = lane >> 4;
  #pragma unroll
  for (int fi = 0; fi < 4; ++fi)
    #pragma unroll
    for (int fj = 0; fj < 2; ++fj)
      #pragma unroll
      for (int r = 0; r < 8; ++r) {
        int lr = wr * 64 + fi * 16 + r + 8 * lhi;
        int lc = wc * 32 + fj * 16 + l16;
        int ch = row0 + lr, n = col0 + lc;
        float v = acc[fi][fj][r] + conv_b[ch];
        y[((long)(b * COUT + ch)) * NPTS + n] = v;
        atomicAdd(&s_sum[lr], v);
        atomicAdd(&s_ssq[lr], v * v);
      }
  __syncthreads();
  if (threadIdx.x < 128) {
    atomicAdd(&sums[row0 + threadIdx.x],  s_sum[threadIdx.x]);
    atomicAdd(&sumsq[row0 + threadIdx.x], s_ssq[threadIdx.x]);
  }
}

// ---------------------------------------------------------------------------
// BN finalize
// ---------------------------------------------------------------------------
__global__ void k_stats(const float* sums, const float* sumsq,
                        const float* gamma, const float* beta,
                        float* scale, float* shift) {
  int c = threadIdx.x;
  float cnt = (float)(B_ * NPTS);
  float mu  = sums[c] / cnt;
  float var = sumsq[c] / cnt - mu * mu;
  float s = gamma[c] * rsqrtf(var + BN_EPS);
  scale[c] = s;
  shift[c] = beta[c] - mu * s;
}

// ---------------------------------------------------------------------------
// Apply BN, emit q^T in bf16 (B,N,256) and per-column 1/||q||
// ---------------------------------------------------------------------------
__global__ void __launch_bounds__(256) k_norm(
    const float* __restrict__ y, const float* scale, const float* shift,
    unsigned short* __restrict__ qbfT, float* __restrict__ rinv) {
  __shared__ float red[256];
  int n = blockIdx.x, b = blockIdx.y, c = threadIdx.x;
  float v = y[((long)(b * COUT + c)) * NPTS + n] * scale[c] + shift[c];
  red[c] = v * v;
  __syncthreads();
  for (int s = 128; s > 0; s >>= 1) {
    if (c < s) red[c] += red[c + s];
    __syncthreads();
  }
  qbfT[((long)(b * NPTS + n)) * COUT + c] = f2bf(v);
  if (c == 0) rinv[b * NPTS + n] = 1.f / fmaxf(sqrtf(red[0]), 1e-12f);
}

// ---------------------------------------------------------------------------
// dist0 = q^T q, raw logits into sim region (symmetric: A and B both from qbfT)
// ---------------------------------------------------------------------------
__global__ void __launch_bounds__(256) k_dist(const unsigned short* __restrict__ qbfT,
                                              float* __restrict__ sim) {
  __shared__ unsigned short At[128][40];
  __shared__ unsigned short Bs[128][40];
  int b = blockIdx.z;
  int row0 = blockIdx.y * 128;   // n tile
  int col0 = blockIdx.x * 128;   // m tile
  v8f acc[4][2];
  #pragma unroll
  for (int i = 0; i < 4; ++i)
    #pragma unroll
    for (int j = 0; j < 2; ++j) acc[i][j] = vzero8();

  const unsigned short* Q = qbfT + (long)b * NPTS * COUT;
  gemm_mainloop(Q, Q, COUT, COUT, COUT, row0, col0, At, Bs, acc);

  int wave = threadIdx.x >> 5, lane = threadIdx.x & 31;
  int wr = wave >> 2, wc = wave & 3, l16 = lane & 15, lhi = lane >> 4;
  #pragma unroll
  for (int fi = 0; fi < 4; ++fi)
    #pragma unroll
    for (int fj = 0; fj < 2; ++fj)
      #pragma unroll
      for (int r = 0; r < 8; ++r) {
        int n = row0 + wr * 64 + fi * 16 + r + 8 * lhi;
        int m = col0 + wc * 32 + fj * 16 + l16;
        sim[((long)b * NPTS + n) * NPTS + m] = acc[fi][fj][r];
      }
}

// ---------------------------------------------------------------------------
// top-9 of raw*rinv_n*rinv_m + rel  (must run before softmax overwrites sim)
// ---------------------------------------------------------------------------
__global__ void __launch_bounds__(256) k_topk(const float* __restrict__ sim,
                                              const float* __restrict__ rinv,
                                              const float* __restrict__ rel,
                                              float* __restrict__ out) {
  int n = blockIdx.x, b = blockIdx.y, tid = threadIdx.x;
  const float* srow = sim + ((long)b * NPTS + n) * NPTS;
  const float* rrow = rel + ((long)b * NPTS + n) * NPTS;
  const float* ri = rinv + b * NPTS;
  float rn = ri[n];

  float bv[KNN]; int bi[KNN];
  #pragma unroll
  for (int i = 0; i < KNN; ++i) { bv[i] = NEG_INF; bi[i] = 0; }
  for (int m = tid; m < NPTS; m += 256) {
    float v = srow[m] * rn * ri[m] + rrow[m];
    if (v > bv[KNN - 1]) {
      int p = KNN - 1;
      while (p > 0 && v > bv[p - 1]) { bv[p] = bv[p - 1]; bi[p] = bi[p - 1]; --p; }
      bv[p] = v; bi[p] = m;
    }
  }

  __shared__ float lv[256 * KNN];
  __shared__ int   li[256 * KNN];
  __shared__ float rv[256];
  __shared__ int   rs[256];
  #pragma unroll
  for (int i = 0; i < KNN; ++i) { lv[tid * KNN + i] = bv[i]; li[tid * KNN + i] = bi[i]; }
  __syncthreads();

  long o_nn = ((long)b * NPTS + n) * KNN;
  long o_ct = (long)B_ * NPTS * KNN + o_nn;
  for (int r = 0; r < KNN; ++r) {
    float mv = NEG_INF; int ms = tid * KNN;
    #pragma unroll
    for (int i = 0; i < KNN; ++i) {
      float v = lv[tid * KNN + i];
      if (v > mv) { mv = v; ms = tid * KNN + i; }
    }
    rv[tid] = mv; rs[tid] = ms;
    __syncthreads();
    for (int s = 128; s > 0; s >>= 1) {
      if (tid < s && rv[tid + s] > rv[tid]) { rv[tid] = rv[tid + s]; rs[tid] = rs[tid + s]; }
      __syncthreads();
    }
    if (tid == 0) {
      int slot = rs[0];
      out[o_nn + r] = (float)li[slot];
      lv[slot] = NEG_INF;
    }
    __syncthreads();
  }
  if (tid < KNN) out[o_ct + tid] = (float)n;
}

// ---------------------------------------------------------------------------
// In-place row softmax over 4096 keys
// ---------------------------------------------------------------------------
__global__ void __launch_bounds__(256) k_softmax(float* __restrict__ sim) {
  __shared__ float red[256];
  int n = blockIdx.x, b = blockIdx.y, tid = threadIdx.x;
  float* row = sim + ((long)b * NPTS + n) * NPTS;
  float vals[16];
  float mx = NEG_INF;
  #pragma unroll
  for (int i = 0; i < 16; ++i) { vals[i] = row[tid + i * 256]; mx = fmaxf(mx, vals[i]); }
  red[tid] = mx; __syncthreads();
  for (int s = 128; s > 0; s >>= 1) { if (tid < s) red[tid] = fmaxf(red[tid], red[tid + s]); __syncthreads(); }
  mx = red[0]; __syncthreads();
  float sm = 0.f;
  #pragma unroll
  for (int i = 0; i < 16; ++i) { vals[i] = __expf(vals[i] - mx); sm += vals[i]; }
  red[tid] = sm; __syncthreads();
  for (int s = 128; s > 0; s >>= 1) { if (tid < s) red[tid] += red[tid + s]; __syncthreads(); }
  float inv = 1.f / red[0];
  #pragma unroll
  for (int i = 0; i < 16; ++i) row[tid + i * 256] = vals[i] * inv;
}

// ---------------------------------------------------------------------------
extern "C" void kernel_launch(void* const* d_in, const int* in_sizes, int n_in,
                              void* d_out, int out_size, void* d_ws, size_t ws_size,
                              hipStream_t stream) {
  const float* x      = (const float*)d_in[0];
  const float* rel    = (const float*)d_in[1];
  const float* conv_w = (const float*)d_in[2];
  const float* conv_b = (const float*)d_in[3];
  const float* gamma  = (const float*)d_in[4];
  const float* beta   = (const float*)d_in[5];
  float* out = (float*)d_out;

  char* ws = (char*)d_ws;
  size_t off = 0;
  auto alloc = [&](size_t bytes) -> void* {
    off = (off + 255) & ~(size_t)255;
    void* p = ws + off;
    off += bytes;
    return p;
  };
  unsigned short* wbf  = (unsigned short*)alloc((size_t)COUT * CIN * 2);
  unsigned short* xbfT = (unsigned short*)alloc((size_t)B_ * NPTS * CIN * 2);
  float* y             = (float*)alloc((size_t)B_ * COUT * NPTS * 4);
  unsigned short* qbfT = (unsigned short*)alloc((size_t)B_ * NPTS * COUT * 2);
  float* rinv          = (float*)alloc((size_t)B_ * NPTS * 4);
  float* sums          = (float*)alloc(COUT * 4);
  float* sumsq         = (float*)alloc(COUT * 4);
  float* scale         = (float*)alloc(COUT * 4);
  float* shift         = (float*)alloc(COUT * 4);

  float* sim = out + (size_t)2 * B_ * NPTS * KNN;   // sim_map region of d_out

  hipLaunchKernelGGL(k_prep_w, dim3((COUT * CIN) / 256), dim3(256), 0, stream,
                     conv_w, wbf, sums, sumsq);
  hipLaunchKernelGGL(k_xT, dim3((B_ * NPTS * CIN) / 256), dim3(256), 0, stream, x, xbfT);
  hipLaunchKernelGGL(k_conv, dim3(NPTS / 128, COUT / 128, B_), dim3(256), 0, stream,
                     wbf, xbfT, conv_b, y, sums, sumsq);
  hipLaunchKernelGGL(k_stats, dim3(1), dim3(COUT), 0, stream,
                     sums, sumsq, gamma, beta, scale, shift);
  hipLaunchKernelGGL(k_norm, dim3(NPTS, B_), dim3(COUT), 0, stream,
                     y, scale, shift, qbfT, rinv);
  hipLaunchKernelGGL(k_dist, dim3(NPTS / 128, NPTS / 128, B_), dim3(256), 0, stream,
                     qbfT, sim);
  hipLaunchKernelGGL(k_topk, dim3(NPTS, B_), dim3(256), 0, stream, sim, rinv, rel, out);
  hipLaunchKernelGGL(k_softmax, dim3(NPTS, B_), dim3(256), 0, stream, sim);
}
